// MoEBlock_55061480735483
// MI455X (gfx1250) — matrix operand order repaired and checked by reading, not compiled, not run on hardware
//
#include <hip/hip_runtime.h>
#include <math.h>

#define DEVI __device__ __forceinline__

typedef __attribute__((ext_vector_type(16))) _Float16 v16h;
typedef __attribute__((ext_vector_type(8)))  float    v8f;

constexpr int Bb = 2, Ss = 2048, Dd = 1024, Hh = 16, Ff = 4096, Ee = 8;
constexpr int NT = Bb * Ss;          // 4096 tokens
constexpr int HD = Dd / Hh;          // 64
constexpr int MAXSLOTS = 2 * NT;     // 8192 (top-2 slots total)

// ---------------------------------------------------------------------------
// CDNA5 async global->LDS copy (ASYNCcnt-tracked), 16B per lane
// ---------------------------------------------------------------------------
DEVI unsigned lds_addr(const void* p) {
  // generic pointers to LDS carry the offset in the low 32 bits (ISA 10.2)
  return (unsigned)(size_t)p;
}
DEVI void async_copy_b128(unsigned lds_off, const void* gaddr) {
  asm volatile("global_load_async_to_lds_b128 %0, %1, off"
               :: "v"(lds_off), "v"(gaddr) : "memory");
}
DEVI void wait_async() { asm volatile("s_wait_asynccnt 0" ::: "memory"); }

// ---------------------------------------------------------------------------
// WMMA helpers: f16 A(16x32) x B(32x16) + f32 C(16x16), wave32 layouts per ISA
// ---------------------------------------------------------------------------
DEVI v8f wmma16(v16h a, v16h b, v8f c) {
  return __builtin_amdgcn_wmma_f32_16x16x32_f16(false, a, false, b, (short)0, c,
                                                false, false);
}

// A fragment: 16x32, row-major tile in LDS (stride in f16 elems).
// lanes 0-15: row=lane, elems 0..7 = K0..7, elems 8..15 = K16..23
// lanes16-31: row=lane-16, K offset +8
DEVI v16h load_fragA(const _Float16* tile, int stride, int row0) {
  const int lane = threadIdx.x & 31;
  const int hl = lane >> 4, r = lane & 15;
  const _Float16* p = tile + (size_t)(row0 + r) * stride;
  union { v16h v; unsigned u[8]; } f;
  const int kb = hl * 8;
#pragma unroll
  for (int j = 0; j < 8; ++j) {
    int kk = ((j < 4) ? (2 * j) : (2 * j + 8)) + kb;
    f.u[j] = *(const unsigned*)(p + kk);
  }
  return f.v;
}

// B fragment: 32x16 stored N-major in LDS as [n][k] (k contiguous, stride elems)
// lanes 0-15: n=lane, elems 0..7 = K0..7, elems 8..15 = K16..23 ; lanes 16-31: n=lane-16, K offset +8
DEVI v16h load_fragB(const _Float16* tile, int stride, int n0) {
  const int lane = threadIdx.x & 31;
  const int hl = lane >> 4, r = lane & 15;
  const _Float16* p = tile + (size_t)(n0 + r) * stride;
  union { v16h v; unsigned u[8]; } f;
  const int kb = hl * 8;
#pragma unroll
  for (int j = 0; j < 8; ++j) f.u[j] = *(const unsigned*)(p + kb + ((j < 4) ? (2 * j) : (2 * j + 8)));
  return f.v;
}

// ---------------------------------------------------------------------------
// Generic 128x128 tiled f16-WMMA GEMM: C = epi(A_f16[M,K] * B_f32[K,N])
// Double-buffered LDS; A tiles via async global->LDS; B tiles converted f32->f16.
// Supports MoE expert batching (blockIdx.y = expert), row gather, slot offset.
// ---------------------------------------------------------------------------
constexpr int EPI_F16 = 0;   // C f16 = acc                       (QKV proj)
constexpr int EPI_RES = 1;   // C f32 = acc + residual; also out2 (O proj)
constexpr int EPI_GELU = 2;  // C f16 = gelu(acc + bias[e])       (MoE up)
constexpr int EPI_BIAS = 3;  // C f32 = acc + bias[e]             (MoE down)

template <int EPI>
__global__ __launch_bounds__(256) void gemm_kernel(
    const _Float16* __restrict__ A, int lda,
    const float* __restrict__ Bw, long strideBe, int ldb,
    const float* __restrict__ bias, int strideBiasE,
    void* __restrict__ Cv, int ldc,
    const float* __restrict__ residual, float* __restrict__ out2,
    const int* __restrict__ rowlist, int maxMe,
    const int* __restrict__ counts, const int* __restrict__ ebase,
    int M, int N, int K) {
  __shared__ _Float16 Al[2][128 * 40];
  __shared__ _Float16 Bl[2][128 * 40];

  const int tid = threadIdx.x;
  const int ntilesN = N >> 7;
  const int e = blockIdx.y;
  const int mt = blockIdx.x / ntilesN;
  const int nt = blockIdx.x % ntilesN;
  const int Me = counts ? counts[e] : M;
  const int m0 = mt * 128;
  if (m0 >= Me) return;  // uniform per block
  const int n0 = nt * 128;
  const int slotbase = ebase ? ebase[e] : 0;
  const float* Bp = Bw + (size_t)e * strideBe;

  const int wave = tid >> 5, lane = tid & 31;
  const int wm = wave >> 1, wn = wave & 1;
  const int hl = lane >> 4, lr = lane & 15;

  // per-thread A-staging geometry (2 threads/row, 16 f16 each)
  const int ar = tid >> 1, aseg = tid & 1;
  const int alocal = m0 + ar;
  const bool avalid = alocal < Me;
  size_t arow;
  if (rowlist)
    arow = avalid ? (size_t)rowlist[(size_t)e * maxMe + alocal] : 0;
  else
    arow = (size_t)(slotbase + (avalid ? alocal : 0));
  const _Float16* asrc0 = A + arow * (size_t)lda + aseg * 16;
  // per-thread B-staging geometry (32 rows x 128 cols, 16 f32 each)
  const int bkr = tid >> 3, bnc = (tid & 7) * 16;

  auto stageA = [&](int k0, _Float16* Ab) {
    _Float16* dst = Ab + ar * 40 + aseg * 16;
    if (avalid) {
      const _Float16* src = asrc0 + k0;
      async_copy_b128(lds_addr(dst), src);
      async_copy_b128(lds_addr(dst + 8), src + 8);
    } else {
      uint4 z = {0, 0, 0, 0};
      ((uint4*)dst)[0] = z;
      ((uint4*)dst)[1] = z;
    }
  };
  auto stageB = [&](int k0, _Float16* Bb) {
    const float* src = Bp + (size_t)(k0 + bkr) * ldb + n0 + bnc;
    if (k0 + 64 < K)  // stream-prefetch 2 k-steps ahead (L2 < w1+w2 footprint)
      __builtin_prefetch(src + (size_t)64 * ldb, 0, 1);
#pragma unroll
    for (int i = 0; i < 16; i += 4) {
      float4 f = *(const float4*)(src + i);
      Bb[(bnc + i + 0) * 40 + bkr] = (_Float16)f.x;
      Bb[(bnc + i + 1) * 40 + bkr] = (_Float16)f.y;
      Bb[(bnc + i + 2) * 40 + bkr] = (_Float16)f.z;
      Bb[(bnc + i + 3) * 40 + bkr] = (_Float16)f.w;
    }
  };

  v8f acc[2][4];
#pragma unroll
  for (int i = 0; i < 2; ++i)
#pragma unroll
    for (int j = 0; j < 4; ++j) acc[i][j] = {};

  stageA(0, Al[0]);
  stageB(0, Bl[0]);
  wait_async();
  __syncthreads();

  const int nk = K >> 5;
  for (int ki = 0; ki < nk; ++ki) {
    const int cur = ki & 1, nxt = cur ^ 1;
    if (ki + 1 < nk) {  // overlap next-tile staging with WMMA on current tile
      stageA((ki + 1) * 32, Al[nxt]);
      stageB((ki + 1) * 32, Bl[nxt]);
    }
    v16h af[2];
    af[0] = load_fragA(Al[cur], 40, wm * 32);
    af[1] = load_fragA(Al[cur], 40, wm * 32 + 16);
    v16h bf[4];
#pragma unroll
    for (int j = 0; j < 4; ++j) bf[j] = load_fragB(Bl[cur], 40, wn * 64 + j * 16);
#pragma unroll
    for (int i = 0; i < 2; ++i)
#pragma unroll
      for (int j = 0; j < 4; ++j) acc[i][j] = wmma16(af[i], bf[j], acc[i][j]);
    wait_async();
    __syncthreads();
  }

  // epilogue: C layout — elem r2 -> row = r2 + 8*(lane>=16), col = lane&15
#pragma unroll
  for (int i = 0; i < 2; ++i) {
#pragma unroll
    for (int j = 0; j < 4; ++j) {
#pragma unroll
      for (int r2 = 0; r2 < 8; ++r2) {
        int rowl = wm * 32 + i * 16 + r2 + hl * 8;
        int coll = wn * 64 + j * 16 + lr;
        if (m0 + rowl >= Me) continue;
        size_t cidx = (size_t)(slotbase + m0 + rowl) * ldc + n0 + coll;
        float v = acc[i][j][r2];
        if (EPI == EPI_F16) {
          ((_Float16*)Cv)[cidx] = (_Float16)v;
        } else if (EPI == EPI_RES) {
          float o = v + residual[(size_t)(m0 + rowl) * ldc + n0 + coll];
          ((float*)Cv)[cidx] = o;
          out2[cidx] = o;
        } else if (EPI == EPI_GELU) {
          float u = v + bias[(size_t)e * strideBiasE + n0 + coll];
          float g = 0.5f * u *
                    (1.0f + tanhf(0.7978845608028654f * (u + 0.044715f * u * u * u)));
          ((_Float16*)Cv)[cidx] = (_Float16)g;
        } else {  // EPI_BIAS
          ((float*)Cv)[cidx] = v + bias[(size_t)e * strideBiasE + n0 + coll];
        }
      }
    }
  }
}

// ---------------------------------------------------------------------------
// Flash attention: block = (b, h, 128-row q tile); 8 waves, 16 q rows each.
// KV tiles of 64, online softmax, WMMA for QK^T and PV. Q/K tiles async->LDS.
// ---------------------------------------------------------------------------
__global__ __launch_bounds__(256) void attn_kernel(const _Float16* __restrict__ q,
                                                   const _Float16* __restrict__ k,
                                                   const _Float16* __restrict__ v,
                                                   _Float16* __restrict__ o) {
  __shared__ _Float16 Ql[128 * 72];
  __shared__ _Float16 Kl[64 * 72];   // [kv][d]   (B layout for QK^T)
  __shared__ _Float16 Vl[64 * 72];   // [d][kv]   (B layout for PV)
  __shared__ _Float16 Pl[8 * 16 * 72];

  const int tid = threadIdx.x;
  const int wave = tid >> 5, lane = tid & 31;
  const int hl = lane >> 4, lr = lane & 15;
  const int qt = blockIdx.x, hh = blockIdx.y, bb = blockIdx.z;
  const size_t rowbase = (size_t)bb * Ss;
  const int cbase = hh * HD;

  {  // async-load Q tile 128x64 once
    int qr = tid >> 1, seg = tid & 1;
    const _Float16* src = q + (rowbase + qt * 128 + qr) * Dd + cbase + seg * 32;
    _Float16* dst = Ql + qr * 72 + seg * 32;
#pragma unroll
    for (int i = 0; i < 4; ++i)
      async_copy_b128(lds_addr(dst + i * 8), src + i * 8);
  }

  v8f O[4];
#pragma unroll
  for (int j = 0; j < 4; ++j) O[j] = {};
  float rowM[8], rowL[8];
#pragma unroll
  for (int r = 0; r < 8; ++r) {
    rowM[r] = -1e30f;
    rowL[r] = 0.f;
  }
  _Float16* Pw = Pl + wave * (16 * 72);

  const int kvtmax = 2 * qt + 1;  // causal limit
  for (int kvt = 0; kvt <= kvtmax; ++kvt) {
    __syncthreads();
    {  // async K tile [kv][d]
      int kr = tid >> 2, c0 = (tid & 3) * 16;
      const _Float16* src = k + (rowbase + kvt * 64 + kr) * Dd + cbase + c0;
      _Float16* dst = Kl + kr * 72 + c0;
      async_copy_b128(lds_addr(dst), src);
      async_copy_b128(lds_addr(dst + 8), src + 8);
    }
    {  // V tile transposed -> [d][kv] (needs per-element shuffle: VALU path)
      int vr = tid >> 2, c0 = (tid & 3) * 16;
      union { uint4 u; _Float16 h[8]; } t0, t1;
      const uint4* src =
          (const uint4*)(v + (rowbase + kvt * 64 + vr) * Dd + cbase + c0);
      t0.u = src[0];
      t1.u = src[1];
#pragma unroll
      for (int i = 0; i < 8; ++i) Vl[(c0 + i) * 72 + vr] = t0.h[i];
#pragma unroll
      for (int i = 0; i < 8; ++i) Vl[(c0 + 8 + i) * 72 + vr] = t1.h[i];
    }
    wait_async();
    __syncthreads();

    // S = Q K^T for this wave's 16 rows x 64 kv
    v8f Sx[4];
#pragma unroll
    for (int j = 0; j < 4; ++j) Sx[j] = {};
    v16h qa0 = load_fragA(Ql, 72, wave * 16);
    v16h qa1 = load_fragA(Ql + 32, 72, wave * 16);
#pragma unroll
    for (int j = 0; j < 4; ++j) {
      Sx[j] = wmma16(qa0, load_fragB(Kl, 72, j * 16), Sx[j]);
      Sx[j] = wmma16(qa1, load_fragB(Kl + 32, 72, j * 16), Sx[j]);
    }

    // causal mask + online softmax (16-lane shuffle reductions per row)
    float pv[4][8], tmax[8];
#pragma unroll
    for (int r = 0; r < 8; ++r) tmax[r] = -1e30f;
#pragma unroll
    for (int j = 0; j < 4; ++j) {
      int scol = kvt * 64 + j * 16 + lr;
#pragma unroll
      for (int r = 0; r < 8; ++r) {
        int srow = qt * 128 + wave * 16 + r + hl * 8;
        float s = Sx[j][r] * 0.125f;  // hd^-0.5
        if (scol > srow) s = -1e30f;
        pv[j][r] = s;
        tmax[r] = fmaxf(tmax[r], s);
      }
    }
#pragma unroll
    for (int r = 0; r < 8; ++r) {
#pragma unroll
      for (int m = 8; m >= 1; m >>= 1)
        tmax[r] = fmaxf(tmax[r], __shfl_xor(tmax[r], m, 32));
      tmax[r] = fmaxf(tmax[r], rowM[r]);
    }
    float scl[8], rsum[8];
#pragma unroll
    for (int r = 0; r < 8; ++r) {
      scl[r] = __expf(rowM[r] - tmax[r]);
      rowM[r] = tmax[r];
      rsum[r] = 0.f;
    }
#pragma unroll
    for (int j = 0; j < 4; ++j) {
#pragma unroll
      for (int r = 0; r < 8; ++r) {
        float p = __expf(pv[j][r] - rowM[r]);
        rsum[r] += p;
        Pw[(r + hl * 8) * 72 + j * 16 + lr] = (_Float16)p;
      }
    }
#pragma unroll
    for (int r = 0; r < 8; ++r) {
#pragma unroll
      for (int m = 8; m >= 1; m >>= 1) rsum[r] += __shfl_xor(rsum[r], m, 32);
      rowL[r] = rowL[r] * scl[r] + rsum[r];
    }
#pragma unroll
    for (int j = 0; j < 4; ++j)
#pragma unroll
      for (int r = 0; r < 8; ++r) O[j][r] *= scl[r];

    // P writes are wave-private: wait LDS stores before fragment reads
    asm volatile("s_wait_dscnt 0" ::: "memory");

    v16h pa0 = load_fragA(Pw, 72, 0);
    v16h pa1 = load_fragA(Pw + 32, 72, 0);
#pragma unroll
    for (int j = 0; j < 4; ++j) {
      O[j] = wmma16(pa0, load_fragB(Vl, 72, j * 16), O[j]);
      O[j] = wmma16(pa1, load_fragB(Vl + 32, 72, j * 16), O[j]);
    }
  }

#pragma unroll
  for (int r = 0; r < 8; ++r) {
    float inv = 1.0f / rowL[r];
    size_t row = rowbase + qt * 128 + wave * 16 + r + hl * 8;
#pragma unroll
    for (int j = 0; j < 4; ++j)
      o[row * Dd + cbase + j * 16 + lr] = (_Float16)(O[j][r] * inv);
  }
}

// ---------------------------------------------------------------------------
// RMSNorm: one token per block, output f16
// ---------------------------------------------------------------------------
__global__ __launch_bounds__(256) void rmsnorm_kernel(const float* __restrict__ x,
                                                      const float* __restrict__ w,
                                                      _Float16* __restrict__ out) {
  __shared__ float red[256];
  const int t = blockIdx.x, tid = threadIdx.x;
  const float* xr = x + (size_t)t * Dd;
  float s = 0.f;
  for (int i = tid; i < Dd; i += 256) {
    float v = xr[i];
    s += v * v;
  }
  red[tid] = s;
  __syncthreads();
  for (int h = 128; h > 0; h >>= 1) {
    if (tid < h) red[tid] += red[tid + h];
    __syncthreads();
  }
  float rs = rsqrtf(red[0] * (1.0f / Dd) + 1e-6f);
  _Float16* orow = out + (size_t)t * Dd;
  for (int i = tid; i < Dd; i += 256) orow[i] = (_Float16)(xr[i] * w[i] * rs);
}

// ---------------------------------------------------------------------------
// Router: one wave per token; logits, softmax, top-2, slot assignment
// ---------------------------------------------------------------------------
__global__ __launch_bounds__(256) void router_kernel(
    const _Float16* __restrict__ h2, const float* __restrict__ rw,
    int* __restrict__ counts, int* __restrict__ rowlist, int* __restrict__ tope,
    int* __restrict__ topp, float* __restrict__ topg) {
  const int tid = threadIdx.x, lane = tid & 31, wave = tid >> 5;
  const int t = blockIdx.x * 8 + wave;
  float acc[Ee];
#pragma unroll
  for (int e = 0; e < Ee; ++e) acc[e] = 0.f;
  const _Float16* hr = h2 + (size_t)t * Dd;
  for (int i = 0; i < Dd / 32; ++i) {
    int d = lane + i * 32;
    float hv = (float)hr[d];
#pragma unroll
    for (int e = 0; e < Ee; ++e) acc[e] += hv * rw[e * Dd + d];
  }
#pragma unroll
  for (int e = 0; e < Ee; ++e)
#pragma unroll
    for (int m = 16; m >= 1; m >>= 1) acc[e] += __shfl_xor(acc[e], m, 32);
  float mx = acc[0];
#pragma unroll
  for (int e = 1; e < Ee; ++e) mx = fmaxf(mx, acc[e]);
  float sum = 0.f, p[Ee];
#pragma unroll
  for (int e = 0; e < Ee; ++e) {
    p[e] = __expf(acc[e] - mx);
    sum += p[e];
  }
  float inv = 1.0f / sum;
#pragma unroll
  for (int e = 0; e < Ee; ++e) p[e] *= inv;
  int i0 = 0;
  float v0 = p[0];
#pragma unroll
  for (int e = 1; e < Ee; ++e)
    if (p[e] > v0) { v0 = p[e]; i0 = e; }
  int i1 = (i0 == 0) ? 1 : 0;
  float v1 = p[i1];
#pragma unroll
  for (int e = 0; e < Ee; ++e)
    if (e != i0 && p[e] > v1) { v1 = p[e]; i1 = e; }
  float gs = 1.0f / (v0 + v1);
  if (lane == 0) {
    int p0 = atomicAdd(&counts[i0], 1);
    int p1 = atomicAdd(&counts[i1], 1);
    rowlist[i0 * NT + p0] = t;
    rowlist[i1 * NT + p1] = t;
    tope[t * 2 + 0] = i0;
    tope[t * 2 + 1] = i1;
    topp[t * 2 + 0] = p0;
    topp[t * 2 + 1] = p1;
    topg[t * 2 + 0] = v0 * gs;
    topg[t * 2 + 1] = v1 * gs;
  }
}

__global__ void zero_counts_kernel(int* counts) {
  if (threadIdx.x < Ee) counts[threadIdx.x] = 0;
}

__global__ void prefix_kernel(const int* __restrict__ counts,
                              int* __restrict__ ebase) {
  if (threadIdx.x == 0) {
    int s = 0;
    for (int e = 0; e < Ee; ++e) {
      ebase[e] = s;
      s += counts[e];
    }
  }
}

// out[t] (= x + attn_out, already stored) += g0*eo[slot0] + g1*eo[slot1]
__global__ __launch_bounds__(256) void combine_kernel(
    float* __restrict__ out, const float* __restrict__ eo,
    const int* __restrict__ tope, const int* __restrict__ topp,
    const float* __restrict__ topg, const int* __restrict__ ebase) {
  const int t = blockIdx.x, tid = threadIdx.x;
  int e0 = tope[t * 2], e1 = tope[t * 2 + 1];
  size_t s0 = (size_t)(ebase[e0] + topp[t * 2]) * Dd;
  size_t s1 = (size_t)(ebase[e1] + topp[t * 2 + 1]) * Dd;
  float g0 = topg[t * 2], g1 = topg[t * 2 + 1];
  float* orow = out + (size_t)t * Dd;
  for (int c = tid; c < Dd; c += 256)
    orow[c] += g0 * eo[s0 + c] + g1 * eo[s1 + c];
}

// ---------------------------------------------------------------------------
extern "C" void kernel_launch(void* const* d_in, const int* in_sizes, int n_in,
                              void* d_out, int out_size, void* d_ws,
                              size_t ws_size, hipStream_t stream) {
  (void)in_sizes; (void)n_in; (void)out_size; (void)ws_size;
  const float* x   = (const float*)d_in[0];
  const float* n1w = (const float*)d_in[1];
  const float* n2w = (const float*)d_in[2];
  const float* wq  = (const float*)d_in[3];
  const float* wk  = (const float*)d_in[4];
  const float* wv  = (const float*)d_in[5];
  const float* wo  = (const float*)d_in[6];
  const float* rw  = (const float*)d_in[7];
  const float* w1  = (const float*)d_in[8];
  const float* b1  = (const float*)d_in[9];
  const float* w2  = (const float*)d_in[10];
  const float* b2  = (const float*)d_in[11];
  float* out = (float*)d_out;
  char* ws = (char*)d_ws;

  size_t off = 0;
  auto alloc = [&](size_t bytes) {
    size_t o = off;
    off += (bytes + 255) & ~(size_t)255;
    return o;
  };
  _Float16* h   = (_Float16*)(ws + alloc((size_t)NT * Dd * 2));
  _Float16* qf  = (_Float16*)(ws + alloc((size_t)NT * Dd * 2));
  _Float16* kf  = (_Float16*)(ws + alloc((size_t)NT * Dd * 2));
  _Float16* vf  = (_Float16*)(ws + alloc((size_t)NT * Dd * 2));
  _Float16* of  = (_Float16*)(ws + alloc((size_t)NT * Dd * 2));
  float*    x2  = (float*)   (ws + alloc((size_t)NT * Dd * 4));
  _Float16* h2  = (_Float16*)(ws + alloc((size_t)NT * Dd * 2));
  _Float16* eh  = (_Float16*)(ws + alloc((size_t)MAXSLOTS * Ff * 2));
  float*    eo  = (float*)   (ws + alloc((size_t)MAXSLOTS * Dd * 4));
  int* counts   = (int*)     (ws + alloc(Ee * 4));
  int* ebase    = (int*)     (ws + alloc(Ee * 4));
  int* rowlist  = (int*)     (ws + alloc((size_t)Ee * NT * 4));
  int* tope     = (int*)     (ws + alloc((size_t)NT * 2 * 4));
  int* topp     = (int*)     (ws + alloc((size_t)NT * 2 * 4));
  float* topg   = (float*)   (ws + alloc((size_t)NT * 2 * 4));

  zero_counts_kernel<<<1, 64, 0, stream>>>(counts);

  // --- attention half ---
  rmsnorm_kernel<<<NT, 256, 0, stream>>>(x, n1w, h);
  dim3 gProj((NT / 128) * (Dd / 128), 1);  // 32*8 = 256 blocks
  gemm_kernel<EPI_F16><<<gProj, 256, 0, stream>>>(
      h, Dd, wq, 0, Dd, nullptr, 0, qf, Dd, nullptr, nullptr, nullptr, 0,
      nullptr, nullptr, NT, Dd, Dd);
  gemm_kernel<EPI_F16><<<gProj, 256, 0, stream>>>(
      h, Dd, wk, 0, Dd, nullptr, 0, kf, Dd, nullptr, nullptr, nullptr, 0,
      nullptr, nullptr, NT, Dd, Dd);
  gemm_kernel<EPI_F16><<<gProj, 256, 0, stream>>>(
      h, Dd, wv, 0, Dd, nullptr, 0, vf, Dd, nullptr, nullptr, nullptr, 0,
      nullptr, nullptr, NT, Dd, Dd);
  attn_kernel<<<dim3(Ss / 128, Hh, Bb), 256, 0, stream>>>(qf, kf, vf, of);
  gemm_kernel<EPI_RES><<<gProj, 256, 0, stream>>>(
      of, Dd, wo, 0, Dd, nullptr, 0, x2, Dd, x, out, nullptr, 0, nullptr,
      nullptr, NT, Dd, Dd);

  // --- MoE half ---
  rmsnorm_kernel<<<NT, 256, 0, stream>>>(x2, n2w, h2);
  router_kernel<<<NT / 8, 256, 0, stream>>>(h2, rw, counts, rowlist, tope, topp,
                                            topg);
  prefix_kernel<<<1, 32, 0, stream>>>(counts, ebase);
  // up-proj + GELU: per-expert gathered rows -> compacted slots (worst-case grid)
  gemm_kernel<EPI_GELU><<<dim3((NT / 128) * (Ff / 128), Ee), 256, 0, stream>>>(
      h2, Dd, w1, (long)Dd * Ff, Ff, b1, Ff, eh, Ff, nullptr, nullptr, rowlist,
      NT, counts, ebase, NT, Ff, Dd);
  // down-proj + bias: slots -> slots
  gemm_kernel<EPI_BIAS><<<dim3((NT / 128) * (Dd / 128), Ee), 256, 0, stream>>>(
      eh, Ff, w2, (long)Ff * Dd, Dd, b2, Dd, eo, Dd, nullptr, nullptr, nullptr,
      NT, counts, ebase, NT, Dd, Ff);
  combine_kernel<<<NT, 256, 0, stream>>>(out, eo, tope, topp, topg, ebase);
}
